// InterestExtractLayer_77790447665310
// MI455X (gfx1250) — matrix operand order repaired and checked by reading, not compiled, not run on hardware
//
#include <hip/hip_runtime.h>
#include <hip/hip_bf16.h>

typedef __attribute__((ext_vector_type(16))) _Float16 v16h;
typedef __attribute__((ext_vector_type(8)))  _Float16 v8h;
typedef __attribute__((ext_vector_type(8)))  float    v8f;

#define T_LEN 200
#define E_DIM 128
#define G_DIM 384
#define WSTR  136   // padded f16 row stride (8-half multiple, bank-conflict free)
#define GSTR  388   // padded f32 gate row stride

// ---- WMMA helpers (CDNA5 16x16x32 f16 -> f32) ------------------------------

__device__ __forceinline__ v8f wmma_f16(v16h a, v16h b, v8f c) {
  // 8 args: (neg_a, A, neg_b, B, c_mod, C, reuse_a, reuse_b)
  return __builtin_amdgcn_wmma_f32_16x16x32_f16(false, a, false, b, (short)0, c,
                                                false, false);
}

// A-matrix 16x32 f16 fragment from row-major LDS tile (ISA 7.12.2 layout):
// lanes 0-15: row M=lane, K = kc+0..7 (VGPR0-3) and kc+16..23 (VGPR4-7)
// lanes 16-31: row M=lane-16, K = kc+8..15 and kc+24..31
__device__ __forceinline__ v16h load_a_frag(const _Float16* base, int lane,
                                            int kc, int stride) {
  const int m   = lane & 15;
  const int off = (lane < 16) ? 0 : 8;
  const _Float16* p = base + m * stride + kc + off;
  v8h lo = *(const v8h*)(p);
  v8h hi = *(const v8h*)(p + 16);
  v16h a;
#pragma unroll
  for (int i = 0; i < 8; ++i) { a[i] = lo[i]; a[i + 8] = hi[i]; }
  return a;
}

// B-matrix 32x16 f16 fragment: lane column N = n0 + (lane&15); lanes 0-15 hold
// K = kc+0..7 and kc+16..23, lanes 16-31 hold K = kc+8..15 and kc+24..31.
// Source tile is N-major (row n, contiguous k) in LDS.
__device__ __forceinline__ v16h load_b_frag(const _Float16* base, int lane,
                                            int n0, int kc, int stride) {
  const int n    = n0 + (lane & 15);
  const int koff = kc + ((lane < 16) ? 0 : 8);
  const _Float16* p = base + n * stride + koff;
  v8h lo = *(const v8h*)(p);
  v8h hi = *(const v8h*)(p + 16);
  v16h b;
#pragma unroll
  for (int i = 0; i < 8; ++i) { b[i] = lo[i]; b[i + 8] = hi[i]; }
  return b;
}

// ---- prep: f32 -> f16 weights (+ transposed, zero-padded MLP weights) ------

__global__ void k_prep(const float* __restrict__ wih, const float* __restrict__ whh,
                       const float* __restrict__ w1,  const float* __restrict__ w2,
                       _Float16* __restrict__ wih_h, _Float16* __restrict__ whh_h,
                       _Float16* __restrict__ w1t,   _Float16* __restrict__ w2t,
                       float* __restrict__ accum) {
  const size_t tid  = (size_t)blockIdx.x * blockDim.x + threadIdx.x;
  const size_t nthr = (size_t)gridDim.x * blockDim.x;
  for (size_t i = tid; i < (size_t)G_DIM * E_DIM; i += nthr) {
    wih_h[i] = (_Float16)wih[i];
    whh_h[i] = (_Float16)whh[i];
  }
  // W1t[n][k] = W1[k][n], padded to 112 rows with zeros (W1 is 256x100)
  for (size_t i = tid; i < (size_t)112 * 256; i += nthr) {
    int n = (int)(i / 256), k = (int)(i % 256);
    w1t[i] = (n < 100) ? (_Float16)w1[k * 100 + n] : (_Float16)0.0f;
  }
  // W2t[n][k] = W2[k][n], padded to 64x128 with zeros (W2 is 100x50)
  for (size_t i = tid; i < (size_t)64 * 128; i += nthr) {
    int n = (int)(i / 128), k = (int)(i % 128);
    w2t[i] = (n < 50 && k < 100) ? (_Float16)w2[k * 50 + n] : (_Float16)0.0f;
  }
  if (tid == 0) *accum = 0.0f;
}

// ---- fused GRU: gi GEMM + gh GEMM + gates, weights resident in 320KB LDS ---

__global__ __launch_bounds__(256) void k_gru(
    const float* __restrict__ ub, const int* __restrict__ blen,
    const _Float16* __restrict__ wih, const _Float16* __restrict__ whh,
    const float* __restrict__ bih, const float* __restrict__ bhh,
    float* __restrict__ out) {
  __shared__ __align__(16) _Float16 sWih[G_DIM * WSTR];  // 104448 B
  __shared__ __align__(16) _Float16 sWhh[G_DIM * WSTR];  // 104448 B
  __shared__ __align__(16) float    sGi[16 * GSTR];      //  24832 B
  __shared__ __align__(16) float    sGh[16 * GSTR];      //  24832 B
  __shared__ __align__(16) _Float16 sXa[16 * WSTR];      //   4352 B
  __shared__ __align__(16) _Float16 sHa[16 * WSTR];      //   4352 B
  __shared__ __align__(16) float    sH[16 * E_DIM];      //   8192 B
  __shared__ float sBih[G_DIM];
  __shared__ float sBhh[G_DIM];
  __shared__ int   sLen[16];

  const int tid  = threadIdx.x;
  const int lane = tid & 31;
  const int wv   = tid >> 5;
  const int b0   = blockIdx.x * 16;

  // stage both weight matrices into padded LDS once
  for (int idx = tid; idx < G_DIM * 16; idx += 256) {
    int row = idx >> 4, c = (idx & 15) * 8;
    *(v8h*)&sWih[row * WSTR + c] = *(const v8h*)&wih[row * E_DIM + c];
    *(v8h*)&sWhh[row * WSTR + c] = *(const v8h*)&whh[row * E_DIM + c];
  }
  for (int g = tid; g < G_DIM; g += 256) { sBih[g] = bih[g]; sBhh[g] = bhh[g]; }
  if (tid < 16) sLen[tid] = blen[b0 + tid];
  for (int i = tid; i < 16 * E_DIM; i += 256) sH[i] = 0.0f;
  for (int i = tid; i < 16 * WSTR; i += 256) sHa[i] = (_Float16)0.0f;
  __syncthreads();

  const int m  = tid >> 4;
  const int e0 = (tid & 15) * 8;
  const size_t rowbase = (size_t)(b0 + m) * T_LEN;

  for (int t = 0; t < T_LEN; ++t) {
    // stage x_t (16 x 128 f32) -> f16 LDS A tile
    {
      const float* px = &ub[(rowbase + t) * E_DIM + e0];
      float4 u = *(const float4*)px;
      float4 v = *(const float4*)(px + 4);
      _Float16* dx = &sXa[m * WSTR + e0];
      dx[0] = (_Float16)u.x; dx[1] = (_Float16)u.y;
      dx[2] = (_Float16)u.z; dx[3] = (_Float16)u.w;
      dx[4] = (_Float16)v.x; dx[5] = (_Float16)v.y;
      dx[6] = (_Float16)v.z; dx[7] = (_Float16)v.w;
      if (t + 1 < T_LEN)
        __builtin_prefetch(&ub[(rowbase + t + 1) * E_DIM + e0], 0, 1);
    }
    __syncthreads();

    // gi = x_t @ W_ih^T ; gh = h @ W_hh^T   (3 N-tiles of 16 per wave)
    v8f acc_i[3] = {};
    v8f acc_h[3] = {};
    for (int kc = 0; kc < E_DIM; kc += 32) {
      v16h ax = load_a_frag(sXa, lane, kc, WSTR);
      v16h ah = load_a_frag(sHa, lane, kc, WSTR);
#pragma unroll
      for (int j = 0; j < 3; ++j) {
        const int n0 = wv * 48 + j * 16;
        v16h bi = load_b_frag(sWih, lane, n0, kc, WSTR);
        acc_i[j] = wmma_f16(ax, bi, acc_i[j]);
        v16h bh = load_b_frag(sWhh, lane, n0, kc, WSTR);
        acc_h[j] = wmma_f16(ah, bh, acc_h[j]);
      }
    }
    // spill gates (+bias) to LDS, C-layout: row = r + (lane<16?0:8), col = lane&15
    {
      const int col = lane & 15;
      const int rb  = (lane < 16) ? 0 : 8;
#pragma unroll
      for (int j = 0; j < 3; ++j) {
        const int g = wv * 48 + j * 16 + col;
        const float bi = sBih[g], bh = sBhh[g];
#pragma unroll
        for (int r = 0; r < 8; ++r) {
          sGi[(r + rb) * GSTR + g] = acc_i[j][r] + bi;
          sGh[(r + rb) * GSTR + g] = acc_h[j][r] + bh;
        }
      }
    }
    __syncthreads();

    // elementwise GRU gate math (8 elems per thread) + output store
    {
      const bool valid = t < sLen[m];
      float o[8];
#pragma unroll
      for (int q = 0; q < 8; ++q) {
        const int j = e0 + q;
        const float ir  = sGi[m * GSTR + j];
        const float hr  = sGh[m * GSTR + j];
        const float iz  = sGi[m * GSTR + 128 + j];
        const float hz  = sGh[m * GSTR + 128 + j];
        const float inn = sGi[m * GSTR + 256 + j];
        const float hnn = sGh[m * GSTR + 256 + j];
        const float rg = 1.0f / (1.0f + __expf(-(ir + hr)));
        const float zg = 1.0f / (1.0f + __expf(-(iz + hz)));
        const float ng = tanhf(inn + rg * hnn);
        const float hc = sH[m * E_DIM + j];
        const float hnew = (1.0f - zg) * ng + zg * hc;
        const float hnx  = valid ? hnew : hc;
        sH[m * E_DIM + j]  = hnx;
        sHa[m * WSTR + j]  = (_Float16)hnx;
        o[q] = valid ? hnew : 0.0f;
      }
      float4* po = (float4*)&out[(rowbase + t) * E_DIM + e0];
      po[0] = make_float4(o[0], o[1], o[2], o[3]);
      po[1] = make_float4(o[4], o[5], o[6], o[7]);
    }
    __syncthreads();
  }
}

// ---- MLP (pos/neg branches) + masked log-loss partial sums -----------------

__global__ __launch_bounds__(256) void k_mlp(
    const float* __restrict__ ub, const float* __restrict__ nub,
    const int* __restrict__ blen, const float* __restrict__ gru,
    const _Float16* __restrict__ w1t, const _Float16* __restrict__ w2t,
    const float* __restrict__ b1, const float* __restrict__ b2,
    const float* __restrict__ w3, const float* __restrict__ pb3,
    float* __restrict__ accum) {
  __shared__ __align__(16) _Float16 sW1[112 * 264];      // 59136 B
  __shared__ __align__(16) _Float16 sW2[64 * 136];       // 17408 B
  __shared__ float sB1[112];
  __shared__ float sB2[64];
  __shared__ float sW3[64];
  __shared__ __align__(16) _Float16 sA[8][16 * 264];     // 67584 B
  __shared__ __align__(16) _Float16 sH1[8][16 * 136];    // 34816 B
  __shared__ float sH2[8][16 * 68];                      // 34816 B
  __shared__ float sAcc;

  const int tid    = threadIdx.x;
  const int lane   = tid & 31;
  const int wv     = tid >> 5;
  const int branch = blockIdx.y;

  for (int idx = tid; idx < 112 * 32; idx += 256) {
    int row = idx >> 5, c = (idx & 31) * 8;
    *(v8h*)&sW1[row * 264 + c] = *(const v8h*)&w1t[row * 256 + c];
  }
  for (int idx = tid; idx < 64 * 16; idx += 256) {
    int row = idx >> 4, c = (idx & 15) * 8;
    *(v8h*)&sW2[row * 136 + c] = *(const v8h*)&w2t[row * 128 + c];
  }
  if (tid < 112) sB1[tid] = (tid < 100) ? b1[tid] : 0.0f;
  if (tid < 64)  sB2[tid] = (tid < 50) ? b2[tid] : 0.0f;
  if (tid >= 64 && tid < 128) { int i = tid - 64; sW3[i] = (i < 50) ? w3[i] : 0.0f; }
  if (tid == 0) sAcc = 0.0f;
  // zero the K=112..135 pad of h1 so layer-2 A fragments read clean zeros
  for (int idx = tid; idx < 8 * 16 * 24; idx += 256) {
    int w = idx / (16 * 24), rem = idx % (16 * 24), r = rem / 24, c = rem % 24;
    sH1[w][r * 136 + 112 + c] = (_Float16)0.0f;
  }
  __syncthreads();

  const float bias3 = *pb3;
  const size_t rows0 = ((size_t)blockIdx.x * 8 + wv) * 16;

  // build A tile: 16 rows x 256 = concat(gru_embed, pos/neg behavior) -> f16
  {
    const int e0 = lane * 8;
    for (int r = 0; r < 16; ++r) {
      const size_t row = rows0 + r;
      const int b = (int)(row / 199), t = (int)(row % 199);
      const float* src;
      if (e0 < 128)
        src = &gru[((size_t)b * 200 + t + 1) * 128 + e0];
      else if (branch == 0)
        src = &ub[((size_t)b * 200 + t + 1) * 128 + (e0 - 128)];
      else
        src = &nub[((size_t)b * 199 + t) * 128 + (e0 - 128)];
      float4 u = *(const float4*)src;
      float4 v = *(const float4*)(src + 4);
      _Float16* d = &sA[wv][r * 264 + e0];
      d[0] = (_Float16)u.x; d[1] = (_Float16)u.y;
      d[2] = (_Float16)u.z; d[3] = (_Float16)u.w;
      d[4] = (_Float16)v.x; d[5] = (_Float16)v.y;
      d[6] = (_Float16)v.z; d[7] = (_Float16)v.w;
    }
  }
  __syncthreads();

  // layer 1: 16x256 @ 256x112 (7 N-tiles, 8 K-chunks)
  v8f a1[7] = {};
  for (int kc = 0; kc < 256; kc += 32) {
    v16h af = load_a_frag(sA[wv], lane, kc, 264);
#pragma unroll
    for (int j = 0; j < 7; ++j) {
      v16h bf = load_b_frag(sW1, lane, j * 16, kc, 264);
      a1[j] = wmma_f16(af, bf, a1[j]);
    }
  }
  {
    const int col = lane & 15, rb = (lane < 16) ? 0 : 8;
#pragma unroll
    for (int j = 0; j < 7; ++j) {
      const int n = j * 16 + col;
      const float bn = sB1[n];
#pragma unroll
      for (int r = 0; r < 8; ++r) {
        float v = fmaxf(a1[j][r] + bn, 0.0f);
        sH1[wv][(r + rb) * 136 + n] = (_Float16)v;
      }
    }
  }
  __syncthreads();

  // layer 2: 16x128 @ 128x64 (4 N-tiles, 4 K-chunks)
  v8f a2[4] = {};
  for (int kc = 0; kc < 128; kc += 32) {
    v16h af = load_a_frag(sH1[wv], lane, kc, 136);
#pragma unroll
    for (int j = 0; j < 4; ++j) {
      v16h bf = load_b_frag(sW2, lane, j * 16, kc, 136);
      a2[j] = wmma_f16(af, bf, a2[j]);
    }
  }
  {
    const int col = lane & 15, rb = (lane < 16) ? 0 : 8;
#pragma unroll
    for (int j = 0; j < 4; ++j) {
      const int n = j * 16 + col;
      const float bn = sB2[n];
#pragma unroll
      for (int r = 0; r < 8; ++r) {
        sH2[wv][(r + rb) * 68 + n] = fmaxf(a2[j][r] + bn, 0.0f);
      }
    }
  }
  __syncthreads();

  // layer 3 (50-wide dot) + sigmoid + clipped log + masked accumulation
  if (lane < 16) {
    const int r = lane;
    const size_t row = rows0 + r;
    const int b = (int)(row / 199), t = (int)(row % 199);
    float s = bias3;
    for (int j = 0; j < 50; ++j) s += sH2[wv][r * 68 + j] * sW3[j];
    const float p = 1.0f / (1.0f + __expf(-s));
    const float term = (branch == 0) ? fmaxf(logf(p), -100.0f)
                                     : fmaxf(logf(1.0f - p), -100.0f);
    if (t < (blen[b] - 1)) atomicAdd(&sAcc, term);
  }
  __syncthreads();
  if (tid == 0) atomicAdd(accum, sAcc);
}

// ---- finalize loss ---------------------------------------------------------

__global__ void k_final(const int* __restrict__ blen,
                        const float* __restrict__ accum,
                        float* __restrict__ loss_out) {
  __shared__ float red[256];
  const int tid = threadIdx.x;
  float s = 0.0f;
  for (int i = tid; i < 2048; i += 256) s += (float)(blen[i] - 1);
  red[tid] = s;
  __syncthreads();
  for (int off = 128; off > 0; off >>= 1) {
    if (tid < off) red[tid] += red[tid + off];
    __syncthreads();
  }
  if (tid == 0) loss_out[0] = -accum[0] / (2.0f * red[0]);
}

// ---- launch ----------------------------------------------------------------

extern "C" void kernel_launch(void* const* d_in, const int* in_sizes, int n_in,
                              void* d_out, int out_size, void* d_ws, size_t ws_size,
                              hipStream_t stream) {
  const float* ub  = (const float*)d_in[0];
  const int*   bl  = (const int*)d_in[1];
  const float* nub = (const float*)d_in[2];
  const float* wih = (const float*)d_in[3];
  const float* whh = (const float*)d_in[4];
  const float* bih = (const float*)d_in[5];
  const float* bhh = (const float*)d_in[6];
  const float* w1  = (const float*)d_in[7];
  const float* b1  = (const float*)d_in[8];
  const float* w2  = (const float*)d_in[9];
  const float* b2  = (const float*)d_in[10];
  const float* w3  = (const float*)d_in[11];
  const float* b3  = (const float*)d_in[12];
  float* out = (float*)d_out;

  char* ws = (char*)d_ws;
  _Float16* wih_h = (_Float16*)(ws + 0);        //  98304 B
  _Float16* whh_h = (_Float16*)(ws + 98304);    //  98304 B
  _Float16* w1t   = (_Float16*)(ws + 196608);   //  57344 B
  _Float16* w2t   = (_Float16*)(ws + 253952);   //  16384 B
  float*    accum = (float*)(ws + 270336);      //      4 B

  k_prep<<<128, 256, 0, stream>>>(wih, whh, w1, w2, wih_h, whh_h, w1t, w2t, accum);
  k_gru<<<128, 256, 0, stream>>>(ub, bl, wih_h, whh_h, bih, bhh, out);
  dim3 gmlp(3184, 2, 1);  // 3184 blocks * 8 waves * 16 rows = 407552 = 2048*199
  k_mlp<<<gmlp, 256, 0, stream>>>(ub, nub, bl, out, w1t, w2t, b1, b2, w3, b3, accum);
  k_final<<<1, 256, 0, stream>>>(bl, accum, out + (size_t)2048 * 200 * 128);
}
